// CedAttention_19816979104362
// MI455X (gfx1250) — matrix operand order repaired and checked by reading, not compiled, not run on hardware
//
#include <hip/hip_runtime.h>
#include <hip/hip_bf16.h>

#define DEVINL __device__ __forceinline__

typedef __attribute__((ext_vector_type(16))) _Float16 v16h;
typedef __attribute__((ext_vector_type(8)))  _Float16 v8h;
typedef __attribute__((ext_vector_type(8)))  float    v8f;
typedef __attribute__((ext_vector_type(4)))  float    v4f;

// Problem constants
constexpr int Bb = 4, Nn = 2048, Cc = 768, Hh = 12, HD = 64;
constexpr int KSTEPS = Cc / 32;  // 24 K-iterations per GEMM
constexpr float L2E = 1.4426950408889634f;

DEVINL v16h hcomb(v8h lo, v8h hi) {
  v16h r;
#pragma unroll
  for (int i = 0; i < 8; ++i) { r[i] = lo[i]; r[i + 8] = hi[i]; }
  return r;
}

DEVINL v8h cvt8(v4f a, v4f b) {
  v8h h;
#pragma unroll
  for (int i = 0; i < 4; ++i) { h[i] = (_Float16)a[i]; h[i + 4] = (_Float16)b[i]; }
  return h;
}

DEVINL v8f wmma_f16(v16h a, v16h b, v8f c) {
  // D = A(16x32 f16) * B(32x16 f16) + C(16x16 f32)
  return __builtin_amdgcn_wmma_f32_16x16x32_f16(false, a, false, b,
                                                (short)0, c, false, false);
}

// A fragment from a row-major [m][k] tile. Lane L: m = L&15,
// elements 0..7 -> K = 8*(L>>4)+e, elements 8..15 -> K = 16+8*(L>>4)+(e-8).
DEVINL v16h frag_a(const _Float16* base, int stride_h, int lane) {
  const int m = lane & 15, g = lane >> 4;
  const _Float16* p = base + m * stride_h + g * 8;
  v8h lo = *(const v8h*)p;
  v8h hi = *(const v8h*)(p + 16);
  return hcomb(lo, hi);
}

// B fragment (KxN tile stored transposed as [n][k]). Lane L: n = L&15,
// elements 0..7 -> K = 8*(L>>4)+e, elements 8..15 -> K = 16+8*(L>>4)+(e-8).
DEVINL v16h frag_b(const _Float16* base, int stride_h, int lane) {
  const int n = lane & 15, g = lane >> 4;
  const _Float16* p = base + n * stride_h + g * 8;
  v8h lo = *(const v8h*)p;
  v8h hi = *(const v8h*)(p + 16);
  return hcomb(lo, hi);
}

// ---------------------------------------------------------------------------
// Kernel 1: qkv = x @ qkv_w^T + qkv_b, scattered into q/k (f16, [B][H][N][64],
// q pre-scaled by HD^-0.5) and v transposed (f16, [B][H][64][N]).
// Block tile 128(M) x 64(N), K-step 32. 8 waves -> 4x2 grid of 32x32 tiles.
// Double-buffered LDS; next tile's global loads overlap current WMMAs.
// ---------------------------------------------------------------------------
__global__ __launch_bounds__(256) void qkv_gemm_kernel(
    const float* __restrict__ x, const float* __restrict__ w,
    const float* __restrict__ bias, _Float16* __restrict__ qws,
    _Float16* __restrict__ kws, _Float16* __restrict__ vtws) {
  constexpr int LDA = 40, LDB = 40;  // halves; rows stay 16B aligned
  __shared__ _Float16 As[2][128 * LDA];
  __shared__ _Float16 Bs[2][64 * LDB];

  const int t = threadIdx.x, lane = t & 31, wid = t >> 5;
  const int wm = wid >> 1, wn = wid & 1;
  const int nBase = blockIdx.x * 64;   // column in [0, 2304)
  const int mBase = blockIdx.y * 128;  // row in [0, 8192)

  v8f acc[2][2] = {};

  const int arow = t >> 1, ako = (t & 1) * 16;
  const int brow = t >> 2, bko = (t & 3) * 8;
  const float* agp = x + (size_t)(mBase + arow) * Cc + ako;
  const float* bgp = w + (size_t)(nBase + brow) * Cc + bko;

  // prologue: tile 0 -> regs -> LDS buf 0
  v4f fa0 = *(const v4f*)(agp + 0), fa1 = *(const v4f*)(agp + 4);
  v4f fa2 = *(const v4f*)(agp + 8), fa3 = *(const v4f*)(agp + 12);
  v4f fb0 = *(const v4f*)(bgp + 0), fb1 = *(const v4f*)(bgp + 4);
  {
    _Float16* sp = As[0] + arow * LDA + ako;
    *(v8h*)(sp) = cvt8(fa0, fa1);
    *(v8h*)(sp + 8) = cvt8(fa2, fa3);
    *(v8h*)(Bs[0] + brow * LDB + bko) = cvt8(fb0, fb1);
  }
  __syncthreads();

  for (int kk = 0; kk < KSTEPS; ++kk) {
    const int cur = kk & 1, nxt = cur ^ 1;
    const bool has_next = (kk + 1 < KSTEPS);
    // issue next tile's global loads early (latency hidden behind WMMAs)
    if (has_next) {
      const int kb = (kk + 1) * 32;
      fa0 = *(const v4f*)(agp + kb + 0);
      fa1 = *(const v4f*)(agp + kb + 4);
      fa2 = *(const v4f*)(agp + kb + 8);
      fa3 = *(const v4f*)(agp + kb + 12);
      fb0 = *(const v4f*)(bgp + kb + 0);
      fb1 = *(const v4f*)(bgp + kb + 4);
    }
    // compute on current buffer
    v16h a0 = frag_a(As[cur] + (wm * 32 + 0) * LDA, LDA, lane);
    v16h a1 = frag_a(As[cur] + (wm * 32 + 16) * LDA, LDA, lane);
    v16h b0 = frag_b(Bs[cur] + (wn * 32 + 0) * LDB, LDB, lane);
    v16h b1 = frag_b(Bs[cur] + (wn * 32 + 16) * LDB, LDB, lane);
    acc[0][0] = wmma_f16(a0, b0, acc[0][0]);
    acc[0][1] = wmma_f16(a0, b1, acc[0][1]);
    acc[1][0] = wmma_f16(a1, b0, acc[1][0]);
    acc[1][1] = wmma_f16(a1, b1, acc[1][1]);
    if (has_next) {
      _Float16* sp = As[nxt] + arow * LDA + ako;
      *(v8h*)(sp) = cvt8(fa0, fa1);
      *(v8h*)(sp + 8) = cvt8(fa2, fa3);
      *(v8h*)(Bs[nxt] + brow * LDB + bko) = cvt8(fb0, fb1);
      __syncthreads();  // uniform condition -> legal
    }
  }

  // This 64-wide column block maps to exactly one of {q,k,v} and one head.
  const int which = nBase / Cc;        // 0=q 1=k 2=v
  const int h = (nBase % Cc) / HD;     // head
  const int b = mBase / Nn;            // batch
  const int g = lane >> 4, ln = lane & 15;

#pragma unroll
  for (int i = 0; i < 2; ++i)
#pragma unroll
    for (int j = 0; j < 2; ++j) {
      const int d = wn * 32 + j * 16 + ln;  // head-dim index in [0,64)
      const float bv = bias[nBase + d];
      const int tok0 = (mBase % Nn) + wm * 32 + i * 16 + g * 8;
      if (which == 2) {
        // transposed V store: 8 consecutive tokens, fixed d -> one b128 store
        v8h pk;
#pragma unroll
        for (int r = 0; r < 8; ++r) pk[r] = (_Float16)(acc[i][j][r] + bv);
        *(v8h*)(vtws + (((size_t)(b * Hh + h)) * HD + d) * Nn + tok0) = pk;
      } else {
        _Float16* dst = (which == 0) ? qws : kws;
        const float s = (which == 0) ? 0.125f : 1.0f;  // fold HD^-0.5 into q
        const size_t rowb = ((size_t)(b * Hh + h)) * Nn;
#pragma unroll
        for (int r = 0; r < 8; ++r)
          dst[(rowb + tok0 + r) * HD + d] = (_Float16)((acc[i][j][r] + bv) * s);
      }
    }
}

// ---------------------------------------------------------------------------
// Kernel 2: causal flash attention. One block per (b, h, 128-query tile).
// 8 waves x 16 query rows. S = Q K^T via WMMA, online softmax, O += P V via
// WMMA with P staged through per-wave LDS and V^T staged through shared LDS.
// ---------------------------------------------------------------------------
__global__ __launch_bounds__(256) void flash_attn_kernel(
    const _Float16* __restrict__ qws, const _Float16* __restrict__ kws,
    const _Float16* __restrict__ vtws, _Float16* __restrict__ ao) {
  constexpr int LVT = 136, LP = 136;     // padded strides (halves)
  __shared__ _Float16 Vt[64 * LVT];      // V^T tile: [d][key]
  __shared__ _Float16 Pb[8 * 16 * LP];   // per-wave P: [row][key]

  const int t = threadIdx.x, lane = t & 31, wv = t >> 5;
  const int qt = blockIdx.x, h = blockIdx.y, b = blockIdx.z;
  const int g = lane >> 4, ln = lane & 15;

  const size_t head = (size_t)b * Hh + h;
  const _Float16* qh = qws + head * Nn * HD;
  const _Float16* kh = kws + head * Nn * HD;
  const _Float16* vh = vtws + head * HD * Nn;

  // Q fragments: wave's 16 rows, 2 K-chunks of 32 over HD=64 (global direct)
  v16h qa[2];
  {
    const _Float16* p = qh + (size_t)(qt * 128 + wv * 16 + ln) * HD;
#pragma unroll
    for (int c = 0; c < 2; ++c) {
      v8h lo = *(const v8h*)(p + c * 32 + g * 8);
      v8h hi = *(const v8h*)(p + c * 32 + 16 + g * 8);
      qa[c] = hcomb(lo, hi);
    }
  }

  float rowmax[8], rowsum[8];
#pragma unroll
  for (int r = 0; r < 8; ++r) { rowmax[r] = -1e30f; rowsum[r] = 0.f; }
  v8f o[4] = {};
  _Float16* Pw = Pb + wv * 16 * LP;

  const int vrow = t >> 2, vko = (t & 3) * 32;

  for (int kt = 0; kt <= qt; ++kt) {  // causal: only key tiles <= query tile
    const int kb = kt * 128;
    // issue this tile's V^T global loads before the barrier (overlaps other
    // waves draining the previous iteration's compute)
    v8h vr[4];
    {
      const _Float16* gp = vh + (size_t)vrow * Nn + kb + vko;
#pragma unroll
      for (int u = 0; u < 4; ++u) vr[u] = *(const v8h*)(gp + u * 8);
    }
    // prefetch next key tile's K rows into cache (global_prefetch path)
    if (kt < qt)
      __builtin_prefetch(kh + (size_t)(kb + 128) * HD + t * 32, 0, 1);
    __syncthreads();  // all waves done reading Vt from previous iteration
    {
      _Float16* sp = Vt + vrow * LVT + vko;
#pragma unroll
      for (int u = 0; u < 4; ++u) *(v8h*)(sp + u * 8) = vr[u];
    }
    __syncthreads();

    // S tile: 16 rows x 128 keys, in eight 16x16 C fragments
    v8f s[8];
    float tmax[8];
#pragma unroll
    for (int r = 0; r < 8; ++r) tmax[r] = -1e30f;
#pragma unroll
    for (int nt = 0; nt < 8; ++nt) {
      v8f sa = {};
      const int key = kb + nt * 16 + ln;
#pragma unroll
      for (int c = 0; c < 2; ++c) {
        // B frag of K^T: lane holds key=n, d values in table order
        const _Float16* p = kh + (size_t)key * HD + c * 32 + g * 8;
        v8h lo = *(const v8h*)p;
        v8h hi = *(const v8h*)(p + 16);
        sa = wmma_f16(qa[c], hcomb(lo, hi), sa);
      }
      if (kt == qt) {  // causal mask on the diagonal tile
#pragma unroll
        for (int r = 0; r < 8; ++r) {
          const int qr = qt * 128 + wv * 16 + r + 8 * g;
          if (key > qr) sa[r] = -1e30f;
        }
      }
#pragma unroll
      for (int r = 0; r < 8; ++r) tmax[r] = fmaxf(tmax[r], sa[r]);
      s[nt] = sa;
    }
    // row max across the 16 lanes holding one row (rows split by half-wave)
#pragma unroll
    for (int r = 0; r < 8; ++r) {
      float v = tmax[r];
      v = fmaxf(v, __shfl_xor(v, 1));
      v = fmaxf(v, __shfl_xor(v, 2));
      v = fmaxf(v, __shfl_xor(v, 4));
      v = fmaxf(v, __shfl_xor(v, 8));
      tmax[r] = v;
    }
    float corr[8];
#pragma unroll
    for (int r = 0; r < 8; ++r) {
      const float nm = fmaxf(rowmax[r], tmax[r]);
      corr[r] = exp2f((rowmax[r] - nm) * L2E);
      rowmax[r] = nm;
      rowsum[r] *= corr[r];  // lane-partial running sum
    }
#pragma unroll
    for (int nto = 0; nto < 4; ++nto)
#pragma unroll
      for (int r = 0; r < 8; ++r) o[nto][r] *= corr[r];

    // P = exp(S - rowmax); stage to per-wave LDS in [row][key] layout
#pragma unroll
    for (int nt = 0; nt < 8; ++nt) {
#pragma unroll
      for (int r = 0; r < 8; ++r) {
        const float p = exp2f((s[nt][r] - rowmax[r]) * L2E);
        rowsum[r] += p;
        Pw[(r + 8 * g) * LP + nt * 16 + ln] = (_Float16)p;
      }
    }

    // O += P @ V : 4 key-chunks x 4 output-dim tiles
#pragma unroll
    for (int c = 0; c < 4; ++c) {
      const _Float16* pp = Pw + (size_t)ln * LP + c * 32 + g * 8;
      v8h plo = *(const v8h*)pp;
      v8h phi = *(const v8h*)(pp + 16);
      const v16h pa = hcomb(plo, phi);
#pragma unroll
      for (int nto = 0; nto < 4; ++nto) {
        const _Float16* vp = Vt + (nto * 16 + ln) * LVT + c * 32 + g * 8;
        v8h bl = *(const v8h*)vp;
        v8h bh = *(const v8h*)(vp + 16);
        o[nto] = wmma_f16(pa, hcomb(bl, bh), o[nto]);
      }
    }
  }

  // finalize: cross-lane sum of lane-partial row sums, normalize, store f16
#pragma unroll
  for (int r = 0; r < 8; ++r) {
    float v = rowsum[r];
    v += __shfl_xor(v, 1);
    v += __shfl_xor(v, 2);
    v += __shfl_xor(v, 4);
    v += __shfl_xor(v, 8);
    rowsum[r] = 1.0f / v;
  }
#pragma unroll
  for (int nto = 0; nto < 4; ++nto) {
    const int d = nto * 16 + ln;
#pragma unroll
    for (int r = 0; r < 8; ++r) {
      const int tok = qt * 128 + wv * 16 + r + 8 * g;
      ao[((size_t)b * Nn + tok) * Cc + h * HD + d] =
          (_Float16)(o[nto][r] * rowsum[r]);
    }
  }
}

// ---------------------------------------------------------------------------
// Kernel 3: out = attn_out @ proj_w^T + proj_b  (f16 A, fp32 out),
// double-buffered like kernel 1.
// ---------------------------------------------------------------------------
__global__ __launch_bounds__(256) void proj_gemm_kernel(
    const _Float16* __restrict__ ain, const float* __restrict__ w,
    const float* __restrict__ bias, float* __restrict__ out) {
  constexpr int LDA = 40, LDB = 40;
  __shared__ _Float16 As[2][128 * LDA];
  __shared__ _Float16 Bs[2][64 * LDB];

  const int t = threadIdx.x, lane = t & 31, wid = t >> 5;
  const int wm = wid >> 1, wn = wid & 1;
  const int nBase = blockIdx.x * 64;
  const int mBase = blockIdx.y * 128;

  v8f acc[2][2] = {};
  const int arow = t >> 1, ako = (t & 1) * 16;
  const int brow = t >> 2, bko = (t & 3) * 8;
  const _Float16* agp = ain + (size_t)(mBase + arow) * Cc + ako;
  const float* bgp = w + (size_t)(nBase + brow) * Cc + bko;

  v8h ha0 = *(const v8h*)(agp), ha1 = *(const v8h*)(agp + 8);
  v4f fb0 = *(const v4f*)(bgp + 0), fb1 = *(const v4f*)(bgp + 4);
  {
    _Float16* sp = As[0] + arow * LDA + ako;
    *(v8h*)(sp) = ha0;
    *(v8h*)(sp + 8) = ha1;
    *(v8h*)(Bs[0] + brow * LDB + bko) = cvt8(fb0, fb1);
  }
  __syncthreads();

  for (int kk = 0; kk < KSTEPS; ++kk) {
    const int cur = kk & 1, nxt = cur ^ 1;
    const bool has_next = (kk + 1 < KSTEPS);
    if (has_next) {
      const int kb = (kk + 1) * 32;
      ha0 = *(const v8h*)(agp + kb);
      ha1 = *(const v8h*)(agp + kb + 8);
      fb0 = *(const v4f*)(bgp + kb + 0);
      fb1 = *(const v4f*)(bgp + kb + 4);
    }
    v16h a0 = frag_a(As[cur] + (wm * 32 + 0) * LDA, LDA, lane);
    v16h a1 = frag_a(As[cur] + (wm * 32 + 16) * LDA, LDA, lane);
    v16h b0 = frag_b(Bs[cur] + (wn * 32 + 0) * LDB, LDB, lane);
    v16h b1 = frag_b(Bs[cur] + (wn * 32 + 16) * LDB, LDB, lane);
    acc[0][0] = wmma_f16(a0, b0, acc[0][0]);
    acc[0][1] = wmma_f16(a0, b1, acc[0][1]);
    acc[1][0] = wmma_f16(a1, b0, acc[1][0]);
    acc[1][1] = wmma_f16(a1, b1, acc[1][1]);
    if (has_next) {
      _Float16* sp = As[nxt] + arow * LDA + ako;
      *(v8h*)(sp) = ha0;
      *(v8h*)(sp + 8) = ha1;
      *(v8h*)(Bs[nxt] + brow * LDB + bko) = cvt8(fb0, fb1);
      __syncthreads();
    }
  }

  const int g = lane >> 4, ln = lane & 15;
#pragma unroll
  for (int i = 0; i < 2; ++i)
#pragma unroll
    for (int j = 0; j < 2; ++j) {
      const int col = nBase + wn * 32 + j * 16 + ln;
      const float bv = bias[col];
      const int row0 = mBase + wm * 32 + i * 16 + g * 8;
#pragma unroll
      for (int r = 0; r < 8; ++r)
        out[(size_t)(row0 + r) * Cc + col] = acc[i][j][r] + bv;
    }
}

// ---------------------------------------------------------------------------
extern "C" void kernel_launch(void* const* d_in, const int* in_sizes, int n_in,
                              void* d_out, int out_size, void* d_ws,
                              size_t ws_size, hipStream_t stream) {
  (void)in_sizes; (void)n_in; (void)out_size; (void)ws_size;
  const float* x      = (const float*)d_in[0];
  const float* qkv_w  = (const float*)d_in[1];
  const float* qkv_b  = (const float*)d_in[2];
  const float* proj_w = (const float*)d_in[3];
  const float* proj_b = (const float*)d_in[4];
  float* out = (float*)d_out;

  // workspace (f16): q, k, v^T, attn_out  -> 4 * B*N*C * 2B = 50.3 MB
  _Float16* ws = (_Float16*)d_ws;
  const size_t per = (size_t)Bb * Hh * Nn * HD;  // == B*N*C elements
  _Float16* qws  = ws;
  _Float16* kws  = ws + per;
  _Float16* vtws = ws + 2 * per;
  _Float16* ao   = ws + 3 * per;

  qkv_gemm_kernel<<<dim3(3 * Cc / 64, (Bb * Nn) / 128), 256, 0, stream>>>(
      x, qkv_w, qkv_b, qws, kws, vtws);
  flash_attn_kernel<<<dim3(Nn / 128, Hh, Bb), 256, 0, stream>>>(
      qws, kws, vtws, ao);
  proj_gemm_kernel<<<dim3(Cc / 64, (Bb * Nn) / 128), 256, 0, stream>>>(
      ao, proj_w, proj_b, out);
}
